// BigramLanguageModel_71811853189891
// MI455X (gfx1250) — compile-verified, matrix operand order checked by reading, not run on hardware
//
#include <hip/hip_runtime.h>
#include <stdint.h>

// ---------------- model dims (compile-time, match reference) ----------------
#define VOCAB   32000
#define NEMB    1024
#define NHEADS  4
#define HEADD   256          // NEMB / NHEADS
#define BATCH   2
#define SEQ     2048
#define NTOK    (BATCH*SEQ)  // 4096
#define FFND    (4*NEMB)     // 4096

// LDS tile geometry: 128 rows x 32 k-elements, row padded to 40 elems (80 B)
// -> 16-lane ds_load_b128 fragment reads hit 16 distinct 4-bank groups.
#define LROW    40

typedef __attribute__((ext_vector_type(16))) __bf16 v16bf;
typedef __attribute__((ext_vector_type(8)))  float  v8f;

union Frag { v16bf v; uint4 q[2]; };

__device__ __forceinline__ __bf16 f2bf(float f) {
    union { float f; uint32_t u; } v; v.f = f;
    uint32_t r = v.u + 0x7FFFu + ((v.u >> 16) & 1u);   // round-to-nearest-even
    uint16_t h = (uint16_t)(r >> 16);
    return __builtin_bit_cast(__bf16, h);
}

// ---- CDNA5 async global->LDS copy (ASYNCcnt-tracked), per-lane 16 bytes ----
__device__ __forceinline__ void async_copy_b128(const void* gptr, void* lptr) {
    unsigned lds  = (unsigned)(uintptr_t)lptr;             // low 32b = LDS offset
    unsigned long long ga = (unsigned long long)(uintptr_t)gptr;
    asm volatile("global_load_async_to_lds_b128 %0, %1, off"
                 :: "v"(lds), "v"(ga) : "memory");
}
__device__ __forceinline__ void wait_async0() {
    asm volatile("s_wait_asynccnt 0x0" ::: "memory");
}

// ---------------------------------------------------------------------------
// bf16 WMMA GEMM with double-buffered async-LDS staging:
//   C[M,N] = epilogue( A[M,K] @ BT[N,K]^T + bias )
// Block = 256 threads = 8 waves (4x2), block tile 128x128, wave tile 32x64.
// Per K-step (32): stage next 128x32 A and B tiles to LDS with
// global_load_async_to_lds_b128 while 8 wmma's consume the current tiles.
// EPI: 0 = f32 store, 1 = bf16 store (+optional ReLU), 2 = bf16 transposed
//      store C[n*ldc+m], 3 = f32 accumulate C[m*ldc+n] += v.
// All dims are multiples of the tile sizes at every call site (no guards).
// ---------------------------------------------------------------------------
template<int EPI>
__global__ __launch_bounds__(256)
void gemm_bf16_wmma(const __bf16* __restrict__ A, int lda,
                    const __bf16* __restrict__ BT, int ldb,
                    void* __restrict__ Cv, int ldc,
                    const float* __restrict__ bias,
                    int M, int N, int K, int relu)
{
    __shared__ __bf16 As[2][128 * LROW];
    __shared__ __bf16 Bs[2][128 * LROW];

    const int tid  = threadIdx.x;
    const int lane = tid & 31;
    const int wave = tid >> 5;               // 0..7
    const int hlf  = lane >> 4;              // 0/1: K-half select per ISA layout
    const int r16  = lane & 15;
    const int wm   = wave >> 1;              // 0..3
    const int wn   = wave & 1;               // 0..1
    const int mblk = blockIdx.y * 128;
    const int nblk = blockIdx.x * 128;

    // staging assignment: 512 16B chunks per tile, 2 per thread
    const int row0 = tid >> 2;               // rows 0..63   (chunk i=0)
    const int row1 = row0 + 64;              // rows 64..127 (chunk i=1)
    const int colE = (tid & 3) * 8;          // element offset of 16B chunk

    auto stage = [&](int k0, int buf) {
        async_copy_b128(A  + (size_t)(mblk + row0) * lda + k0 + colE,
                        &As[buf][row0 * LROW + colE]);
        async_copy_b128(A  + (size_t)(mblk + row1) * lda + k0 + colE,
                        &As[buf][row1 * LROW + colE]);
        async_copy_b128(BT + (size_t)(nblk + row0) * ldb + k0 + colE,
                        &Bs[buf][row0 * LROW + colE]);
        async_copy_b128(BT + (size_t)(nblk + row1) * ldb + k0 + colE,
                        &Bs[buf][row1 * LROW + colE]);
    };

    v8f acc[2][4] = {};

    const int nk = K >> 5;
    stage(0, 0);
    for (int kt = 0; kt < nk; ++kt) {
        const int cur = kt & 1;
        wait_async0();          // my async stores to LDS complete
        __syncthreads();        // -> everyone's are complete & prior reads done
        if (kt + 1 < nk) stage((kt + 1) << 5, cur ^ 1);

        Frag a[2], b[4];
        #pragma unroll
        for (int mi = 0; mi < 2; ++mi) {
            const __bf16* p = &As[cur][(wm*32 + mi*16 + r16) * LROW + 8*hlf];
            a[mi].q[0] = *reinterpret_cast<const uint4*>(p);
            a[mi].q[1] = *reinterpret_cast<const uint4*>(p + 16);
        }
        #pragma unroll
        for (int ni = 0; ni < 4; ++ni) {
            const __bf16* p = &Bs[cur][(wn*64 + ni*16 + r16) * LROW + 8*hlf];
            b[ni].q[0] = *reinterpret_cast<const uint4*>(p);
            b[ni].q[1] = *reinterpret_cast<const uint4*>(p + 16);
        }
        #pragma unroll
        for (int mi = 0; mi < 2; ++mi)
            #pragma unroll
            for (int ni = 0; ni < 4; ++ni)
                acc[mi][ni] = __builtin_amdgcn_wmma_f32_16x16x32_bf16(
                    false, a[mi].v, false, b[ni].v,
                    (short)0, acc[mi][ni], false, false);
    }

    const int m0 = mblk + wm * 32;
    const int n0 = nblk + wn * 64;
    #pragma unroll
    for (int mi = 0; mi < 2; ++mi) {
        #pragma unroll
        for (int ni = 0; ni < 4; ++ni) {
            const int n = n0 + ni*16 + r16;
            const float bs = bias ? bias[n] : 0.0f;
            #pragma unroll
            for (int j = 0; j < 8; ++j) {
                const int m = m0 + mi*16 + 8*hlf + j;   // C/D lane layout
                float v = acc[mi][ni][j] + bs;
                if (EPI == 1 && relu) v = fmaxf(v, 0.0f);
                if constexpr (EPI == 0) {
                    ((float*)Cv)[(size_t)m * ldc + n] = v;
                } else if constexpr (EPI == 1) {
                    ((__bf16*)Cv)[(size_t)m * ldc + n] = f2bf(v);
                } else if constexpr (EPI == 2) {
                    ((__bf16*)Cv)[(size_t)n * ldc + m] = f2bf(v);
                } else {
                    float* C = (float*)Cv;
                    C[(size_t)m * ldc + n] += v;
                }
            }
        }
    }
}

// ------------------------ embedding: x = tok_emb[idx] + pos_emb -------------
__global__ void embed_kernel(const int* __restrict__ idx,
                             const float* __restrict__ tok,
                             const float* __restrict__ pos,
                             float* __restrict__ x)
{
    const int token = blockIdx.x;            // 0..NTOK-1
    const int t     = token % SEQ;
    const int id    = idx[token];
    const float* tr = tok + (size_t)id * NEMB;
    const float* pr = pos + (size_t)t  * NEMB;
    float* xr = x + (size_t)token * NEMB;
    for (int c = threadIdx.x; c < NEMB; c += 256)
        xr[c] = tr[c] + pr[c];
}

// --------------- layernorm over last dim, output bf16 -----------------------
__global__ void layernorm_bf16(const float* __restrict__ X,
                               const float* __restrict__ g,
                               const float* __restrict__ b,
                               __bf16* __restrict__ Y)
{
    const int row = blockIdx.x, tid = threadIdx.x;
    __shared__ float r0[256], r1[256];
    const float* xr = X + (size_t)row * NEMB;
    float s = 0.f, sq = 0.f;
    for (int c = tid; c < NEMB; c += 256) { float v = xr[c]; s += v; sq += v*v; }
    r0[tid] = s; r1[tid] = sq; __syncthreads();
    for (int st = 128; st > 0; st >>= 1) {
        if (tid < st) { r0[tid] += r0[tid+st]; r1[tid] += r1[tid+st]; }
        __syncthreads();
    }
    const float mu  = r0[0] * (1.0f/NEMB);
    const float var = r1[0] * (1.0f/NEMB) - mu*mu;
    const float rstd = rsqrtf(var + 1e-5f);
    __bf16* yr = Y + (size_t)row * NEMB;
    for (int c = tid; c < NEMB; c += 256)
        yr[c] = f2bf((xr[c] - mu) * rstd * g[c] + b[c]);
}

// ------------- fp32 [K,N] -> bf16 transposed [N,K] (LDS tiled) --------------
__global__ void cvt_transpose(const float* __restrict__ W,
                              __bf16* __restrict__ WT, int K, int N)
{
    __shared__ float tile[32][33];
    const int k0 = blockIdx.y * 32, n0 = blockIdx.x * 32;
    for (int i = threadIdx.y; i < 32; i += 8)
        tile[i][threadIdx.x] = W[(size_t)(k0 + i) * N + n0 + threadIdx.x];
    __syncthreads();
    for (int i = threadIdx.y; i < 32; i += 8)
        WT[(size_t)(n0 + i) * K + k0 + threadIdx.x] = f2bf(tile[threadIdx.x][i]);
}

// ------------------------- elementwise f32 -> bf16 --------------------------
__global__ void cvt_bf16(const float* __restrict__ X, __bf16* __restrict__ Y,
                         size_t n)
{
    for (size_t i = (size_t)blockIdx.x * blockDim.x + threadIdx.x; i < n;
         i += (size_t)gridDim.x * blockDim.x)
        Y[i] = f2bf(X[i]);
}

// -------- causal softmax of one score row, scaled, bf16 output --------------
__global__ void softmax_causal(const float* __restrict__ S,
                               __bf16* __restrict__ P, float scale)
{
    const int row = blockIdx.x, tid = threadIdx.x;
    __shared__ float red[256];
    const float* sr = S + (size_t)row * SEQ;
    __bf16* pr = P + (size_t)row * SEQ;
    float mx = -3.0e38f;
    for (int c = tid; c <= row; c += 256) mx = fmaxf(mx, sr[c] * scale);
    red[tid] = mx; __syncthreads();
    for (int st = 128; st > 0; st >>= 1) {
        if (tid < st) red[tid] = fmaxf(red[tid], red[tid+st]);
        __syncthreads();
    }
    mx = red[0]; __syncthreads();
    float sum = 0.f;
    for (int c = tid; c <= row; c += 256) sum += expf(sr[c] * scale - mx);
    red[tid] = sum; __syncthreads();
    for (int st = 128; st > 0; st >>= 1) {
        if (tid < st) red[tid] += red[tid+st];
        __syncthreads();
    }
    const float inv = 1.0f / red[0];
    for (int c = tid; c < SEQ; c += 256)
        pr[c] = f2bf((c <= row) ? expf(sr[c] * scale - mx) * inv : 0.0f);
}

// ---------------- per-token NLL over the vocab ------------------------------
__global__ void nll_rows(const float* __restrict__ logits,
                         const int* __restrict__ tgt,
                         float* __restrict__ nll)
{
    const int row = blockIdx.x, tid = threadIdx.x;
    __shared__ float red[256];
    const float* lr = logits + (size_t)row * VOCAB;
    float mx = -3.0e38f;
    for (int c = tid; c < VOCAB; c += 256) mx = fmaxf(mx, lr[c]);
    red[tid] = mx; __syncthreads();
    for (int st = 128; st > 0; st >>= 1) {
        if (tid < st) red[tid] = fmaxf(red[tid], red[tid+st]);
        __syncthreads();
    }
    mx = red[0]; __syncthreads();
    float sum = 0.f;
    for (int c = tid; c < VOCAB; c += 256) sum += expf(lr[c] - mx);
    red[tid] = sum; __syncthreads();
    for (int st = 128; st > 0; st >>= 1) {
        if (tid < st) red[tid] += red[tid+st];
        __syncthreads();
    }
    if (tid == 0) nll[row] = mx + logf(red[0]) - lr[tgt[row]];
}

// ------------- deterministic single-block mean reduction --------------------
__global__ void reduce_mean(const float* __restrict__ nll, float* __restrict__ out)
{
    const int tid = threadIdx.x;
    __shared__ float red[256];
    float s = 0.f;
    for (int i = tid; i < NTOK; i += 256) s += nll[i];
    red[tid] = s; __syncthreads();
    for (int st = 128; st > 0; st >>= 1) {
        if (tid < st) red[tid] += red[tid+st];
        __syncthreads();
    }
    if (tid == 0) out[0] = red[0] * (1.0f / NTOK);
}

// ---------------------------------------------------------------------------
extern "C" void kernel_launch(void* const* d_in, const int* in_sizes, int n_in,
                              void* d_out, int out_size, void* d_ws, size_t ws_size,
                              hipStream_t stream)
{
    (void)in_sizes; (void)n_in; (void)out_size; (void)ws_size;
    const int*   idx     = (const int*)  d_in[0];
    const int*   targets = (const int*)  d_in[1];
    const float* tok_emb = (const float*)d_in[2];
    const float* pos_emb = (const float*)d_in[3];
    const float* wq      = (const float*)d_in[4];
    const float* wk      = (const float*)d_in[5];
    const float* wv      = (const float*)d_in[6];
    const float* ln1_g   = (const float*)d_in[7];
    const float* ln1_b   = (const float*)d_in[8];
    const float* ln2_g   = (const float*)d_in[9];
    const float* ln2_b   = (const float*)d_in[10];
    const float* w1      = (const float*)d_in[11];
    const float* b1      = (const float*)d_in[12];
    const float* w2      = (const float*)d_in[13];
    const float* b2      = (const float*)d_in[14];
    const float* lm_w    = (const float*)d_in[15];
    const float* lm_b    = (const float*)d_in[16];

    float* logits = (float*)d_out;                       // [NTOK, VOCAB]
    float* loss   = logits + (size_t)NTOK * VOCAB;       // single f32

    // -------- workspace carve-out (256B aligned slabs) --------
    char* wp = (char*)d_ws;
    auto alloc = [&](size_t bytes) {
        char* p = wp; wp += (bytes + 255) & ~(size_t)255; return (void*)p;
    };
    float*  x    = (float*) alloc((size_t)NTOK * NEMB * 4);   // residual stream
    __bf16* hb   = (__bf16*)alloc((size_t)NTOK * NEMB * 2);   // LN output
    __bf16* qb   = (__bf16*)alloc((size_t)NHEADS * NTOK * HEADD * 2);
    __bf16* kb   = (__bf16*)alloc((size_t)NHEADS * NTOK * HEADD * 2);
    __bf16* vT   = (__bf16*)alloc((size_t)NHEADS * HEADD * NTOK * 2); // V^T
    __bf16* wqT  = (__bf16*)alloc((size_t)NHEADS * HEADD * NEMB * 2);
    __bf16* wkT  = (__bf16*)alloc((size_t)NHEADS * HEADD * NEMB * 2);
    __bf16* wvT  = (__bf16*)alloc((size_t)NHEADS * HEADD * NEMB * 2);
    __bf16* w1T  = (__bf16*)alloc((size_t)FFND * NEMB * 2);
    __bf16* w2T  = (__bf16*)alloc((size_t)NEMB * FFND * 2);
    __bf16* lmT  = (__bf16*)alloc((size_t)VOCAB * NEMB * 2);
    float*  S    = (float*) alloc((size_t)SEQ * SEQ * 4);
    __bf16* P    = (__bf16*)alloc((size_t)SEQ * SEQ * 2);
    __bf16* f1   = (__bf16*)alloc((size_t)NTOK * FFND * 2);
    __bf16* xb   = (__bf16*)alloc((size_t)NTOK * NEMB * 2);
    float*  nll  = (float*) alloc((size_t)NTOK * 4);

    const dim3 tb(32, 8);

    // -------- weight prep: f32 -> bf16, transposed to [N,K] --------
    for (int h = 0; h < NHEADS; ++h) {
        cvt_transpose<<<dim3(HEADD/32, NEMB/32), tb, 0, stream>>>(
            wq + (size_t)h*NEMB*HEADD, wqT + (size_t)h*HEADD*NEMB, NEMB, HEADD);
        cvt_transpose<<<dim3(HEADD/32, NEMB/32), tb, 0, stream>>>(
            wk + (size_t)h*NEMB*HEADD, wkT + (size_t)h*HEADD*NEMB, NEMB, HEADD);
        cvt_transpose<<<dim3(HEADD/32, NEMB/32), tb, 0, stream>>>(
            wv + (size_t)h*NEMB*HEADD, wvT + (size_t)h*HEADD*NEMB, NEMB, HEADD);
    }
    cvt_transpose<<<dim3(FFND/32,  NEMB/32), tb, 0, stream>>>(w1,  w1T, NEMB, FFND);
    cvt_transpose<<<dim3(NEMB/32,  FFND/32), tb, 0, stream>>>(w2,  w2T, FFND, NEMB);
    cvt_transpose<<<dim3(VOCAB/32, NEMB/32), tb, 0, stream>>>(lm_w, lmT, NEMB, VOCAB);

    // -------- embedding + LN1 --------
    embed_kernel<<<NTOK, 256, 0, stream>>>(idx, tok_emb, pos_emb, x);
    layernorm_bf16<<<NTOK, 256, 0, stream>>>(x, ln1_g, ln1_b, hb);

    // -------- QKV projections (per head; V stored transposed) --------
    for (int h = 0; h < NHEADS; ++h) {
        gemm_bf16_wmma<1><<<dim3(HEADD/128, NTOK/128), 256, 0, stream>>>(
            hb, NEMB, wqT + (size_t)h*HEADD*NEMB, NEMB,
            qb + (size_t)h*NTOK*HEADD, HEADD, nullptr, NTOK, HEADD, NEMB, 0);
        gemm_bf16_wmma<1><<<dim3(HEADD/128, NTOK/128), 256, 0, stream>>>(
            hb, NEMB, wkT + (size_t)h*HEADD*NEMB, NEMB,
            kb + (size_t)h*NTOK*HEADD, HEADD, nullptr, NTOK, HEADD, NEMB, 0);
        gemm_bf16_wmma<2><<<dim3(HEADD/128, NTOK/128), 256, 0, stream>>>(
            hb, NEMB, wvT + (size_t)h*HEADD*NEMB, NEMB,
            vT + (size_t)h*HEADD*NTOK, NTOK, nullptr, NTOK, HEADD, NEMB, 0);
    }

    // -------- attention per (batch, head); scale = NEMB^-0.5 = 1/32 --------
    const float scale = 0.03125f;
    for (int bb = 0; bb < BATCH; ++bb) {
        for (int h = 0; h < NHEADS; ++h) {
            const __bf16* Qs = qb + ((size_t)h*NTOK + (size_t)bb*SEQ) * HEADD;
            const __bf16* Ks = kb + ((size_t)h*NTOK + (size_t)bb*SEQ) * HEADD;
            // S = Q @ K^T   (K itself is BT in [N=SEQ, K=HEADD] form)
            gemm_bf16_wmma<0><<<dim3(SEQ/128, SEQ/128), 256, 0, stream>>>(
                Qs, HEADD, Ks, HEADD, S, SEQ, nullptr, SEQ, SEQ, HEADD, 0);
            softmax_causal<<<SEQ, 256, 0, stream>>>(S, P, scale);
            // x[b, :, h*HEADD:] += P @ V   (vT is BT in [N=HEADD, K=SEQ] form)
            gemm_bf16_wmma<3><<<dim3(HEADD/128, SEQ/128), 256, 0, stream>>>(
                P, SEQ, vT + (size_t)h*HEADD*NTOK + (size_t)bb*SEQ, NTOK,
                x + (size_t)bb*SEQ*NEMB + (size_t)h*HEADD, NEMB,
                nullptr, SEQ, HEADD, SEQ, 0);
        }
    }

    // -------- FFN: x += relu(LN2(x) @ w1 + b1) @ w2 + b2 --------
    layernorm_bf16<<<NTOK, 256, 0, stream>>>(x, ln2_g, ln2_b, hb);
    gemm_bf16_wmma<1><<<dim3(FFND/128, NTOK/128), 256, 0, stream>>>(
        hb, NEMB, w1T, NEMB, f1, FFND, b1, NTOK, FFND, NEMB, 1);
    gemm_bf16_wmma<3><<<dim3(NEMB/128, NTOK/128), 256, 0, stream>>>(
        f1, FFND, w2T, FFND, x, NEMB, b2, NTOK, NEMB, FFND, 0);

    // -------- LM head + loss --------
    cvt_bf16<<<4096, 256, 0, stream>>>(x, xb, (size_t)NTOK * NEMB);
    gemm_bf16_wmma<0><<<dim3(VOCAB/128, NTOK/128), 256, 0, stream>>>(
        xb, NEMB, lmT, NEMB, logits, VOCAB, lm_b, NTOK, VOCAB, NEMB, 0);
    nll_rows<<<NTOK, 256, 0, stream>>>(logits, targets, nll);
    reduce_mean<<<1, 256, 0, stream>>>(nll, loss);
}
